// ResidualAttentionBlock_15083925144167
// MI455X (gfx1250) — matrix operand order repaired and checked by reading, not compiled, not run on hardware
//
#include <hip/hip_runtime.h>
#include <hip/hip_bf16.h>

typedef __attribute__((ext_vector_type(16))) _Float16 v16h;
typedef __attribute__((ext_vector_type(8)))  _Float16 v8h;
typedef __attribute__((ext_vector_type(8)))  float    v8f;

#define CH     128
#define NPOS   4096
#define NB     16
#define NGRP   8
#define EPS_GN 1e-5f

__device__ __forceinline__ v16h cat8(v8h lo, v8h hi) {
    return __builtin_shufflevector(lo, hi, 0, 1, 2, 3, 4, 5, 6, 7,
                                   8, 9, 10, 11, 12, 13, 14, 15);
}

// A-fragment (16x32 f16, ISA 7.12.2): lane row m, halves 0..7 = K 8h..8h+7,
// halves 8..15 = K 16+8h..23+8h.  `arow` = row base (row-major, stride>=32).
__device__ __forceinline__ v16h fragA(const _Float16* arow, int lhalf) {
    v8h lo = *(const v8h*)(arow + 8 * lhalf);
    v8h hi = *(const v8h*)(arow + 16 + 8 * lhalf);
    return cat8(lo, hi);
}

// B-fragment (32x16 f16): lane column n, halves 0..7 = K 8h..8h+7, halves 8..15 = K 16+8h..23+8h.
// `bcol` = column base of column-major tile (K contiguous, stride>=32).
__device__ __forceinline__ v16h fragB(const _Float16* bcol, int lhalf) {
    v8h lo = *(const v8h*)(bcol + 8 * lhalf);
    v8h hi = *(const v8h*)(bcol + 16 + 8 * lhalf);
    return cat8(lo, hi);
}

// Async global->LDS copy of 16 bytes per lane (ASYNCcnt; bypasses VGPRs).
__device__ __forceinline__ void async_cp16(void* lds_dst, const void* gsrc) {
    unsigned l = (unsigned)(unsigned long long)lds_dst;   // LDS byte offset
    asm volatile("global_load_async_to_lds_b128 %0, %1, off"
                 :: "v"(l), "v"(gsrc) : "memory");
}
__device__ __forceinline__ void async_wait0() {
    asm volatile("s_wait_asynccnt 0x0" ::: "memory");
}
// Compiler-only ordering for per-wave LDS RAW (hardware keeps DS in order).
__device__ __forceinline__ void lds_order() {
    asm volatile("" ::: "memory");
}

// ---------------------------------------------------------------------------
// Kernel 0: convert weights to f16 once (scale folded into wq).
// ---------------------------------------------------------------------------
__global__ __launch_bounds__(256)
void wcvt_kernel(const float* __restrict__ wq, const float* __restrict__ wk,
                 const float* __restrict__ wv, const float* __restrict__ wf,
                 _Float16* __restrict__ w16) {
    const int i = blockIdx.x * 256 + threadIdx.x;         // 0..16383
    const float sc = 0.08838834764831845f;                // 128^-0.5
    w16[i]             = (_Float16)(wq[i] * sc);
    w16[16384 + i]     = (_Float16)wk[i];
    w16[2 * 16384 + i] = (_Float16)wv[i];
    w16[3 * 16384 + i] = (_Float16)wf[i];
}

// ---------------------------------------------------------------------------
// Kernel 1: GroupNorm -> f16 activations xgn[b][c][n]
// ---------------------------------------------------------------------------
__global__ __launch_bounds__(256)
void gn_kernel(const float* __restrict__ x, const float* __restrict__ gamma,
               const float* __restrict__ beta, _Float16* __restrict__ xgn) {
    const int b = blockIdx.x / NGRP, g = blockIdx.x % NGRP;
    const int cpg = CH / NGRP;                            // 16
    const long base = ((long)b * CH + (long)g * cpg) * NPOS;
    const int len = cpg * NPOS;                           // 65536

    __shared__ float rs[256], rq[256];
    float s = 0.f, q = 0.f;
    for (int i = threadIdx.x; i < len; i += 256) {
        float v = x[base + i];
        s += v; q += v * v;
    }
    rs[threadIdx.x] = s; rq[threadIdx.x] = q;
    __syncthreads();
    for (int off = 128; off > 0; off >>= 1) {
        if (threadIdx.x < off) {
            rs[threadIdx.x] += rs[threadIdx.x + off];
            rq[threadIdx.x] += rq[threadIdx.x + off];
        }
        __syncthreads();
    }
    __shared__ float smean, sinv;
    if (threadIdx.x == 0) {
        float mean = rs[0] / (float)len;
        float var  = rq[0] / (float)len - mean * mean;
        smean = mean;
        sinv  = rsqrtf(var + EPS_GN);
    }
    __syncthreads();
    const float mean = smean, inv = sinv;
    for (int i = threadIdx.x; i < len; i += 256) {
        int c = g * cpg + (i >> 12);
        float v = (x[base + i] - mean) * inv * gamma[c] + beta[c];
        xgn[base + i] = (_Float16)v;
    }
}

// ---------------------------------------------------------------------------
// Kernel 2: Q/K/V projections. Each B fragment is loaded once and feeds
// three WMMAs (q,k,v).
// ---------------------------------------------------------------------------
__global__ __launch_bounds__(256)
void qkv_kernel(const _Float16* __restrict__ xgn,
                const _Float16* __restrict__ w16,
                const float* __restrict__ bq, const float* __restrict__ bk,
                const float* __restrict__ bv,
                _Float16* __restrict__ qT, _Float16* __restrict__ kT,
                _Float16* __restrict__ vbuf) {
    const int b  = blockIdx.x / (NPOS / 64);
    const int n0 = (blockIdx.x % (NPOS / 64)) * 64;

    __shared__ alignas(16) _Float16 xs[64 * CH];          // 16 KB
    for (int idx = threadIdx.x; idx < 1024; idx += 256) {
        int c = idx >> 3, np = idx & 7;
        v8h t = *(const v8h*)(xgn + ((long)(b * CH + c)) * NPOS + n0 + np * 8);
        #pragma unroll
        for (int u = 0; u < 8; ++u) xs[(np * 8 + u) * CH + c] = t[u];
    }
    __syncthreads();

    const int wave = threadIdx.x >> 5;
    const int lane = threadIdx.x & 31;
    const int lhalf = lane >> 4, ln = lane & 15;
    const int m0 = wave * 16;
    const float scale = 0.08838834764831845f;

    const _Float16* wq16 = w16;
    const _Float16* wk16 = w16 + 16384;
    const _Float16* wv16 = w16 + 2 * 16384;

    v8f accq[4], acck[4], accv[4];
    #pragma unroll
    for (int t = 0; t < 4; ++t)
        #pragma unroll
        for (int i = 0; i < 8; ++i) {
            accq[t][i] = 0.f; acck[t][i] = 0.f; accv[t][i] = 0.f;
        }

    #pragma unroll
    for (int cs = 0; cs < 4; ++cs) {
        v16h aq = fragA(wq16 + (m0 + ln) * CH + cs * 32, lhalf);
        v16h ak = fragA(wk16 + (m0 + ln) * CH + cs * 32, lhalf);
        v16h av = fragA(wv16 + (m0 + ln) * CH + cs * 32, lhalf);
        #pragma unroll
        for (int nt = 0; nt < 4; ++nt) {
            v16h bm = fragB(xs + (nt * 16 + ln) * CH + cs * 32, lhalf);
            accq[nt] = __builtin_amdgcn_wmma_f32_16x16x32_f16(
                false, aq, false, bm, (short)0, accq[nt], false, false);
            acck[nt] = __builtin_amdgcn_wmma_f32_16x16x32_f16(
                false, ak, false, bm, (short)0, acck[nt], false, false);
            accv[nt] = __builtin_amdgcn_wmma_f32_16x16x32_f16(
                false, av, false, bm, (short)0, accv[nt], false, false);
        }
    }
    #pragma unroll
    for (int nt = 0; nt < 4; ++nt) {
        const int n = n0 + nt * 16 + ln;
        #pragma unroll
        for (int i = 0; i < 8; ++i) {
            const int m = m0 + i + 8 * lhalf;
            qT[((long)b * NPOS + n) * CH + m]     = (_Float16)(accq[nt][i] + bq[m] * scale);
            kT[((long)b * NPOS + n) * CH + m]     = (_Float16)(acck[nt][i] + bk[m]);
            vbuf[((long)(b * CH + m)) * NPOS + n] = (_Float16)(accv[nt][i] + bv[m]);
        }
    }
}

// ---------------------------------------------------------------------------
// Kernel 3: flash attention, j-step 64, double-buffered async K/V staging.
// ---------------------------------------------------------------------------
__device__ __forceinline__ void stage_tiles(_Float16* ksb, _Float16* vsb,
                                            const _Float16* kTsrc,
                                            const _Float16* vbase,
                                            int j0, int tid) {
    // K tile: kT rows j0..j0+63, contiguous 8192 halves -> [j][c]
    #pragma unroll
    for (int t = 0; t < 4; ++t) {
        int idx = tid + t * 256;
        async_cp16(ksb + idx * 8, kTsrc + idx * 8);
    }
    // V tile: 128 rows of 64 halves -> [c][j]
    #pragma unroll
    for (int t = 0; t < 4; ++t) {
        int chunk = tid + t * 256;
        int c = chunk >> 3, part = chunk & 7;
        async_cp16(vsb + c * 64 + part * 8,
                   vbase + (long)c * NPOS + j0 + part * 8);
    }
}

__global__ __launch_bounds__(256)
void attn_kernel(const _Float16* __restrict__ qT,
                 const _Float16* __restrict__ kT,
                 const _Float16* __restrict__ vbuf,
                 _Float16* __restrict__ oT) {
    const int b    = blockIdx.x / (NPOS / 128);
    const int iblk = blockIdx.x % (NPOS / 128);
    const int wave = threadIdx.x >> 5;
    const int lane = threadIdx.x & 31;
    const int lhalf = lane >> 4, ln = lane & 15;
    const int i0 = iblk * 128 + wave * 16;
    const int tid = threadIdx.x;

    __shared__ alignas(16) _Float16 ks2[2][64 * CH];      // 32 KB
    __shared__ alignas(16) _Float16 vs2[2][CH * 64];      // 32 KB
    __shared__ alignas(16) _Float16 ps[8][16 * 64];       // 16 KB

    const _Float16* vbase = vbuf + (long)b * CH * NPOS;

    const _Float16* qrow = qT + ((long)b * NPOS + i0 + ln) * CH;
    v16h aQ[4];
    #pragma unroll
    for (int cs = 0; cs < 4; ++cs) aQ[cs] = fragA(qrow + cs * 32, lhalf);

    v8f oacc[8];
    #pragma unroll
    for (int t = 0; t < 8; ++t)
        #pragma unroll
        for (int i = 0; i < 8; ++i) oacc[t][i] = 0.f;
    float mrun[8], lrun[8];
    #pragma unroll
    for (int i = 0; i < 8; ++i) { mrun[i] = -1e30f; lrun[i] = 0.f; }

    int cur = 0;
    stage_tiles(ks2[0], vs2[0], kT + (long)b * NPOS * CH, vbase, 0, tid);
    async_wait0();
    __syncthreads();

    for (int jb = 0; jb < NPOS / 64; ++jb) {
        const int j0 = jb * 64;
        if (jb + 1 < NPOS / 64)       // prefetch next tile while computing
            stage_tiles(ks2[cur ^ 1], vs2[cur ^ 1],
                        kT + ((long)b * NPOS + j0 + 64) * CH, vbase, j0 + 64, tid);
        const _Float16* ksb = ks2[cur];
        const _Float16* vsb = vs2[cur];

        // 4 S tiles over 64 j-columns
        v8f s[4];
        #pragma unroll
        for (int jt = 0; jt < 4; ++jt)
            #pragma unroll
            for (int i = 0; i < 8; ++i) s[jt][i] = 0.f;
        #pragma unroll
        for (int cs = 0; cs < 4; ++cs)
            #pragma unroll
            for (int jt = 0; jt < 4; ++jt) {
                v16h bk_ = fragB(ksb + (jt * 16 + ln) * CH + cs * 32, lhalf);
                s[jt] = __builtin_amdgcn_wmma_f32_16x16x32_f16(
                    false, aQ[cs], false, bk_, (short)0, s[jt], false, false);
            }

        // Online softmax over 64 columns
        float alpha[8];
        #pragma unroll
        for (int i = 0; i < 8; ++i) {
            float rm = fmaxf(fmaxf(s[0][i], s[1][i]), fmaxf(s[2][i], s[3][i]));
            #pragma unroll
            for (int msk = 1; msk < 16; msk <<= 1)
                rm = fmaxf(rm, __shfl_xor(rm, msk, 32));
            float mnew = fmaxf(mrun[i], rm);
            alpha[i] = __expf(mrun[i] - mnew);
            float rsum = 0.f;
            #pragma unroll
            for (int jt = 0; jt < 4; ++jt) {
                float p = __expf(s[jt][i] - mnew);
                s[jt][i] = p;
                rsum += p;
            }
            #pragma unroll
            for (int msk = 1; msk < 16; msk <<= 1)
                rsum += __shfl_xor(rsum, msk, 32);
            lrun[i] = lrun[i] * alpha[i] + rsum;
            mrun[i] = mnew;
        }

        // Stage P row-major [i][64]; per-wave buffer, DS in-order per wave.
        _Float16* pw = ps[wave];
        #pragma unroll
        for (int i = 0; i < 8; ++i) {
            int m = i + 8 * lhalf;
            #pragma unroll
            for (int jt = 0; jt < 4; ++jt)
                pw[m * 64 + jt * 16 + ln] = (_Float16)s[jt][i];
        }
        lds_order();

        v16h aP0 = fragA(pw + ln * 64, lhalf);
        v16h aP1 = fragA(pw + ln * 64 + 32, lhalf);
        #pragma unroll
        for (int ct = 0; ct < 8; ++ct) {
            v16h bv0 = fragB(vsb + (ct * 16 + ln) * 64, lhalf);
            v16h bv1 = fragB(vsb + (ct * 16 + ln) * 64 + 32, lhalf);
            #pragma unroll
            for (int i = 0; i < 8; ++i) oacc[ct][i] *= alpha[i];
            oacc[ct] = __builtin_amdgcn_wmma_f32_16x16x32_f16(
                false, aP0, false, bv0, (short)0, oacc[ct], false, false);
            oacc[ct] = __builtin_amdgcn_wmma_f32_16x16x32_f16(
                false, aP1, false, bv1, (short)0, oacc[ct], false, false);
        }

        async_wait0();        // next tile's copies landed (mine)
        __syncthreads();      // everyone done reading cur + copies visible
        cur ^= 1;
    }

    // Normalize and write O^T [b][i][c] f16
    float rl[8];
    #pragma unroll
    for (int i = 0; i < 8; ++i) rl[i] = __builtin_amdgcn_rcpf(lrun[i]);
    #pragma unroll
    for (int ct = 0; ct < 8; ++ct)
        #pragma unroll
        for (int i = 0; i < 8; ++i) {
            int m = i + 8 * lhalf;
            float v = oacc[ct][i] * rl[i];
            oT[((long)b * NPOS + i0 + m) * CH + ct * 16 + ln] = (_Float16)v;
        }
}

// ---------------------------------------------------------------------------
// Kernel 4: output projection + bias + residual (f32 output).
// ---------------------------------------------------------------------------
__global__ __launch_bounds__(256)
void proj_kernel(const _Float16* __restrict__ oT,
                 const _Float16* __restrict__ wf16, const float* __restrict__ bf,
                 const float* __restrict__ x, float* __restrict__ out) {
    const int b  = blockIdx.x / (NPOS / 64);
    const int n0 = (blockIdx.x % (NPOS / 64)) * 64;

    __shared__ alignas(16) _Float16 os[64 * CH];          // 16 KB
    {
        const _Float16* src = oT + ((long)b * NPOS + n0) * CH;
        #pragma unroll
        for (int t = 0; t < 4; ++t) {
            int idx = threadIdx.x + t * 256;
            async_cp16(os + idx * 8, src + idx * 8);
        }
    }
    async_wait0();
    __syncthreads();

    const int wave = threadIdx.x >> 5;
    const int lane = threadIdx.x & 31;
    const int lhalf = lane >> 4, ln = lane & 15;
    const int m0 = wave * 16;

    v8f acc[4];
    #pragma unroll
    for (int t = 0; t < 4; ++t)
        #pragma unroll
        for (int i = 0; i < 8; ++i) acc[t][i] = 0.f;

    #pragma unroll
    for (int cs = 0; cs < 4; ++cs) {
        v16h a = fragA(wf16 + (m0 + ln) * CH + cs * 32, lhalf);
        #pragma unroll
        for (int nt = 0; nt < 4; ++nt) {
            v16h bm = fragB(os + (nt * 16 + ln) * CH + cs * 32, lhalf);
            acc[nt] = __builtin_amdgcn_wmma_f32_16x16x32_f16(
                false, a, false, bm, (short)0, acc[nt], false, false);
        }
    }
    #pragma unroll
    for (int nt = 0; nt < 4; ++nt) {
        const int n = n0 + nt * 16 + ln;
        #pragma unroll
        for (int i = 0; i < 8; ++i) {
            const int m = m0 + i + 8 * lhalf;
            const long addr = ((long)(b * CH + m)) * NPOS + n;
            out[addr] = acc[nt][i] + bf[m] + x[addr];
        }
    }
}

// ---------------------------------------------------------------------------
extern "C" void kernel_launch(void* const* d_in, const int* in_sizes, int n_in,
                              void* d_out, int out_size, void* d_ws, size_t ws_size,
                              hipStream_t stream) {
    const float* x    = (const float*)d_in[0];
    const float* gn_w = (const float*)d_in[1];
    const float* gn_b = (const float*)d_in[2];
    const float* wq   = (const float*)d_in[3];
    const float* bq   = (const float*)d_in[4];
    const float* wk   = (const float*)d_in[5];
    const float* bk   = (const float*)d_in[6];
    const float* wv   = (const float*)d_in[7];
    const float* bv   = (const float*)d_in[8];
    const float* wf   = (const float*)d_in[9];
    const float* bf   = (const float*)d_in[10];
    float* out = (float*)d_out;

    _Float16* ws = (_Float16*)d_ws;
    const size_t SZ = (size_t)NB * CH * NPOS;             // 8,388,608 halves
    _Float16* xgn = ws;                                   // reused as oT
    _Float16* qT  = ws + SZ;
    _Float16* kT  = ws + 2 * SZ;
    _Float16* vb  = ws + 3 * SZ;
    _Float16* w16 = ws + 4 * SZ;                          // 4 x 16384 halves
    _Float16* oT  = xgn;

    wcvt_kernel<<<64,                256, 0, stream>>>(wq, wk, wv, wf, w16);
    gn_kernel  <<<NB * NGRP,         256, 0, stream>>>(x, gn_w, gn_b, xgn);
    qkv_kernel <<<NB * (NPOS / 64),  256, 0, stream>>>(xgn, w16, bq, bk, bv,
                                                       qT, kT, vb);
    attn_kernel<<<NB * (NPOS / 128), 256, 0, stream>>>(qT, kT, vb, oT);
    proj_kernel<<<NB * (NPOS / 64),  256, 0, stream>>>(oT, w16 + 3 * 16384, bf,
                                                       x, out);
}
